// AdaptiveRecurrentModel_54116587930227
// MI455X (gfx1250) — matrix operand order repaired and checked by reading, not compiled, not run on hardware
//
#include <hip/hip_runtime.h>
#include <hip/hip_bf16.h>
#include <math.h>

// ---------------------------------------------------------------------------
// AdaptiveRecurrentModel for MI455X (gfx1250, wave32, WMMA + async-to-LDS)
//
//   h_{t+1}[b,g] = tanh( (x@Wi^T)[b,g] + (h@Wsum^T)[b,g]
//                        + sum_k (h[b,k/128]*x[b,k%128]) * WaR[g,k] + bsum[g] )
//   WaR[g,k] = Wa[g*128 + k/128, k%128]   (contiguous 16384-element row)
//   Wsum = W_base + ba.reshape(128,128),  bsum = bi + bias
//
// 8 waves (256 threads), one persistent workgroup; each wave owns a 32x32
// output block (2x2 register tiling -> 4 f32 accumulators). x_{t+1} is
// prefetched into LDS with global_load_async_to_lds_b128, overlapped with the
// 2080 WMMAs of step t and drained with s_wait_asynccnt.
// ---------------------------------------------------------------------------

typedef __attribute__((ext_vector_type(16))) __bf16 v16bf;
typedef __attribute__((ext_vector_type(8)))  __bf16 v8bf;
typedef __attribute__((ext_vector_type(8)))  float  v8f;

#define TT 256
#define BB 64
#define DD 128
#define HH 128
#define KBIG (HH * DD)      // 16384
#define HROW 136            // padded bf16 stride for h rows (272B = 17*16B)

static __device__ inline v16bf cat8(v8bf lo, v8bf hi) {
  return __builtin_shufflevector(lo, hi, 0,1,2,3,4,5,6,7,8,9,10,11,12,13,14,15);
}

// A-matrix fragment, 16x32 bf16 (ISA 7.12.2): lanes<16 hold K {0..7,16..23},
// lanes>=16 hold K {8..15,24..31}, row M = lane&15.
static __device__ inline v16bf afrag(const __bf16* row, int k0, int half) {
  const __bf16* p = row + k0 + half * 8;
  return cat8(*(const v8bf*)p, *(const v8bf*)(p + 16));
}

// Scale a cached A-fragment by the recurrent-state scalar (h[b,hi] * x[b,d]).
static __device__ inline v16bf scalef(v16bf x, float s) {
  v16bf a;
#pragma unroll
  for (int i = 0; i < 16; i++) a[i] = (__bf16)((float)x[i] * s);
  return a;
}

// B-matrix fragment, 32x16 bf16: column n = lane&15, same K order as A:
// lanes<16 cover K {0..7,16..23}, lanes>=16 cover K {8..15,24..31}. rowptr = &WaR[g,0].
static __device__ inline v16bf bfrag(const __bf16* rowptr, int k0, int khalf) {
  const __bf16* q = rowptr + k0 + khalf * 8;
  return cat8(*(const v8bf*)q, *(const v8bf*)(q + 16));
}

static __device__ inline v8f wmma_bf16(v16bf a, v16bf b, v8f c) {
  return __builtin_amdgcn_wmma_f32_16x16x32_bf16(false, a, false, b, (short)0, c,
                                                 false, false);
}

// Async copy of one token step (64x128 bf16 = 16KB = 1024 x 16B chunks) into
// LDS, 4 chunks per thread, tracked by ASYNCcnt.
static __device__ inline void issue_x_async(const __bf16* __restrict__ tokB,
                                            int t, __bf16* dst, int tid) {
  const __bf16* src = tokB + (size_t)t * (BB * DD);
#pragma unroll
  for (int j = 0; j < 4; j++) {
    const int c = j * 256 + tid;                    // 16B-chunk index 0..1023
    unsigned ldsa = (unsigned)(size_t)(dst + c * 8);
    unsigned long long ga = (unsigned long long)(size_t)(src + c * 8);
    asm volatile("global_load_async_to_lds_b128 %0, %1, off"
                 :: "v"(ldsa), "v"(ga) : "memory");
  }
}

// ---------------------------------------------------------------------------
// Pre-pass: bf16 conversions (Wa, tokens, Wi), Wsum = W_base + ba,
// bsum = bi + bias.  Note T*B*D == H*H*D == 2097152.
// ---------------------------------------------------------------------------
__global__ void arm_prep_kernel(const float* __restrict__ tok,
                                const float* __restrict__ Wa,
                                const float* __restrict__ Wi,
                                const float* __restrict__ Wb,
                                const float* __restrict__ ba,
                                const float* __restrict__ bi,
                                const float* __restrict__ bias,
                                __bf16* __restrict__ tokB,
                                __bf16* __restrict__ WaB,
                                __bf16* __restrict__ WiB,
                                __bf16* __restrict__ WsB,
                                float* __restrict__ bsum) {
  int i = blockIdx.x * blockDim.x + threadIdx.x;
  if (i < HH * KBIG) {
    WaB[i]  = (__bf16)Wa[i];
    tokB[i] = (__bf16)tok[i];
  }
  if (i < HH * HH) {
    WiB[i] = (__bf16)Wi[i];
    WsB[i] = (__bf16)(Wb[i] + ba[i]);
  }
  if (i < HH) bsum[i] = bi[i] + bias[i];
}

// ---------------------------------------------------------------------------
__global__ __launch_bounds__(256, 1)
void arm_rnn_kernel(const __bf16* __restrict__ tokB,
                    const __bf16* __restrict__ WaB,
                    const __bf16* __restrict__ WiB,
                    const __bf16* __restrict__ WsB,
                    const float* __restrict__ bsum,
                    float* __restrict__ out) {
  __shared__ __align__(16) __bf16 xs[2][BB * DD];   // x double buffer (linear)
  __shared__ __align__(16) __bf16 hs[BB * HROW];    // h, padded stride

  const int tid   = threadIdx.x;
  const int lane  = tid & 31;
  const int wave  = tid >> 5;          // 0..7
  const int mt2   = wave >> 2;         // b-block  (0..1) -> rows 32*mt2..+31
  const int nt2   = wave & 3;          // g-block  (0..3) -> cols 32*nt2..+31
  const int ln    = lane & 15;
  const int half  = lane >> 4;
  const int brow0 = mt2 * 32 + ln, brow1 = brow0 + 16;   // A rows / out rows
  const int grow0 = nt2 * 32 + ln, grow1 = grow0 + 16;   // B rows / out cols

  // h0 = 0
  for (int i = tid; i < BB * HROW; i += 256) hs[i] = (__bf16)0.0f;

  // stage x_0
  issue_x_async(tokB, 0, &xs[0][0], tid);
  asm volatile("s_wait_asynccnt 0x0" ::: "memory");
  __syncthreads();

#pragma unroll 1
  for (int t = 0; t < TT; t++) {
    const int cb = t & 1;
    if (t + 1 < TT) issue_x_async(tokB, t + 1, &xs[cb ^ 1][0], tid);

    const __bf16* x0 = &xs[cb][brow0 * DD];
    const __bf16* x1 = &xs[cb][brow1 * DD];
    const __bf16* h0 = &hs[brow0 * HROW];
    const __bf16* h1 = &hs[brow1 * HROW];

    // cache the 4+4 x fragments for the whole step (reused 129x each)
    v16bf xf0[4], xf1[4];
#pragma unroll
    for (int dk = 0; dk < 4; dk++) {
      xf0[dk] = afrag(x0, dk * 32, half);
      xf1[dk] = afrag(x1, dk * 32, half);
    }

    v8f a00 = {0.f,0.f,0.f,0.f,0.f,0.f,0.f,0.f};
    v8f a01 = {0.f,0.f,0.f,0.f,0.f,0.f,0.f,0.f};
    v8f a10 = {0.f,0.f,0.f,0.f,0.f,0.f,0.f,0.f};
    v8f a11 = {0.f,0.f,0.f,0.f,0.f,0.f,0.f,0.f};

    // ---- x @ Wi^T (K = 128) ----
    const __bf16* wi0 = WiB + grow0 * DD;
    const __bf16* wi1 = WiB + grow1 * DD;
#pragma unroll
    for (int dk = 0; dk < 4; dk++) {
      v16bf b0 = bfrag(wi0, dk * 32, half);
      v16bf b1 = bfrag(wi1, dk * 32, half);
      a00 = wmma_bf16(xf0[dk], b0, a00);
      a01 = wmma_bf16(xf0[dk], b1, a01);
      a10 = wmma_bf16(xf1[dk], b0, a10);
      a11 = wmma_bf16(xf1[dk], b1, a11);
    }

    // ---- h @ Wsum^T (K = 128; carries W_base and ba) ----
    const __bf16* ws0 = WsB + grow0 * HH;
    const __bf16* ws1 = WsB + grow1 * HH;
#pragma unroll
    for (int dk = 0; dk < 4; dk++) {
      v16bf af0 = afrag(h0, dk * 32, half);
      v16bf af1 = afrag(h1, dk * 32, half);
      v16bf b0  = bfrag(ws0, dk * 32, half);
      v16bf b1  = bfrag(ws1, dk * 32, half);
      a00 = wmma_bf16(af0, b0, a00);
      a01 = wmma_bf16(af0, b1, a01);
      a10 = wmma_bf16(af1, b0, a10);
      a11 = wmma_bf16(af1, b1, a11);
    }

    // ---- bilinear GEMM: K = 16384, A = outer(h_row, x_row) on the fly ----
    const __bf16* wa0 = WaB + (size_t)grow0 * KBIG;
    const __bf16* wa1 = WaB + (size_t)grow1 * KBIG;
#pragma unroll 1
    for (int hi = 0; hi < HH; hi++) {
      const float s0 = (float)h0[hi];
      const float s1 = (float)h1[hi];
      if (hi + 1 < HH) __builtin_prefetch(wa0 + (size_t)(hi + 1) * DD, 0, 3);
#pragma unroll
      for (int dk = 0; dk < 4; dk++) {
        const int k0 = hi * DD + dk * 32;
        v16bf b0 = bfrag(wa0, k0, half);
        v16bf b1 = bfrag(wa1, k0, half);
        v16bf a0 = scalef(xf0[dk], s0);
        v16bf a1 = scalef(xf1[dk], s1);
        a00 = wmma_bf16(a0, b0, a00);
        a01 = wmma_bf16(a0, b1, a01);
        a10 = wmma_bf16(a1, b0, a10);
        a11 = wmma_bf16(a1, b1, a11);
      }
    }

    // ---- tanh into registers, then single-buffer h update ----
    const float bs0 = bsum[grow0];
    const float bs1 = bsum[grow1];
#pragma unroll
    for (int r = 0; r < 8; r++) {
      a00[r] = tanhf(a00[r] + bs0);
      a01[r] = tanhf(a01[r] + bs1);
      a10[r] = tanhf(a10[r] + bs0);
      a11[r] = tanhf(a11[r] + bs1);
    }

    __syncthreads();              // every wave finished reading hs
#pragma unroll
    for (int r = 0; r < 8; r++) {
      const int m = r + half * 8;
      hs[(mt2 * 32 + m)      * HROW + grow0] = (__bf16)a00[r];
      hs[(mt2 * 32 + m)      * HROW + grow1] = (__bf16)a01[r];
      hs[(mt2 * 32 + 16 + m) * HROW + grow0] = (__bf16)a10[r];
      hs[(mt2 * 32 + 16 + m) * HROW + grow1] = (__bf16)a11[r];
    }
    asm volatile("s_wait_asynccnt 0x0" ::: "memory");  // x_{t+1} landed in LDS
    __syncthreads();              // h writes + async x visible to all waves
  }

  // ---- emit h_final [64,128] f32 ----
  {
    const int b = tid >> 2, d0 = (tid & 3) * 32;
#pragma unroll
    for (int j = 0; j < 32; j++)
      out[b * HH + d0 + j] = (float)hs[b * HROW + d0 + j];
  }
}

// ---------------------------------------------------------------------------
extern "C" void kernel_launch(void* const* d_in, const int* in_sizes, int n_in,
                              void* d_out, int out_size, void* d_ws, size_t ws_size,
                              hipStream_t stream) {
  const float* tokens = (const float*)d_in[0];
  const float* Wi     = (const float*)d_in[1];
  const float* bi     = (const float*)d_in[2];
  const float* W_base = (const float*)d_in[3];
  const float* bias   = (const float*)d_in[4];
  const float* Wa     = (const float*)d_in[5];
  const float* ba     = (const float*)d_in[6];
  float* out = (float*)d_out;

  char* ws = (char*)d_ws;
  __bf16* WaB  = (__bf16*)(ws);                       // 4,194,304 B
  __bf16* tokB = (__bf16*)(ws + 4194304);             // 4,194,304 B
  __bf16* WiB  = (__bf16*)(ws + 8388608);             //    32,768 B
  __bf16* WsB  = (__bf16*)(ws + 8421376);             //    32,768 B
  float*  bsum = (float*)(ws + 8454144);              //       512 B

  arm_prep_kernel<<<(HH * KBIG + 255) / 256, 256, 0, stream>>>(
      tokens, Wa, Wi, W_base, ba, bi, bias, tokB, WaB, WiB, WsB, bsum);

  arm_rnn_kernel<<<1, 256, 0, stream>>>(tokB, WaB, WiB, WsB, bsum, out);
}
